// DecoderLayer_68204080660773
// MI455X (gfx1250) — matrix operand order repaired and checked by reading, not compiled, not run on hardware
//
#include <hip/hip_runtime.h>
#include <hip/hip_bf16.h>

// ---------------------------------------------------------------------------
// Transformer decoder layer for MI455X (gfx1250), wave32 + WMMA f16->f32.
// All GEMMs run as C[M,N] = A[M,K](f16) x BT[N,K](f16) via
// v_wmma_f32_16x16x32_f16; both operands are K-contiguous so every fragment
// load is a pair of 16B global loads matching the ISA VGPR layouts.
// K-loop is manually double-buffered (unrolled x2, requires K % 64 == 0,
// true for all launches here: K in {64, 1024, 4096}) so global loads stay in
// flight underneath the WMMA issue stream, plus global_prefetch two tiles
// ahead. Workspace requirement: ~320 MB.
// ---------------------------------------------------------------------------

typedef _Float16 f16;
typedef _Float16 v16h __attribute__((ext_vector_type(16)));
typedef _Float16 v8h  __attribute__((ext_vector_type(8)));
typedef float    v8f  __attribute__((ext_vector_type(8)));

__device__ __forceinline__ v16h mk16(v8h lo, v8h hi) {
  return __builtin_shufflevector(lo, hi, 0,1,2,3,4,5,6,7,8,9,10,11,12,13,14,15);
}

__device__ __forceinline__ float warp_sum(float v) {
#pragma unroll
  for (int o = 16; o; o >>= 1) v += __shfl_xor(v, o);
  return v;
}
__device__ __forceinline__ float warp_max(float v) {
#pragma unroll
  for (int o = 16; o; o >>= 1) v = fmaxf(v, __shfl_xor(v, o));
  return v;
}

// Load one k-step of fragments: A (16x32 layout: K-halves 0..7 / 16..23, +8
// for upper half-wave) and B from BT (N x K: same K layout as A).
__device__ __forceinline__ void load_frags(const f16* const aRow[2],
                                           const f16* const bRow[4],
                                           int k, v16h a[2], v16h b[4]) {
#pragma unroll
  for (int i = 0; i < 2; ++i)
    a[i] = mk16(*(const v8h*)(aRow[i] + k), *(const v8h*)(aRow[i] + k + 16));
#pragma unroll
  for (int j = 0; j < 4; ++j)
    b[j] = mk16(*(const v8h*)(bRow[j] + k), *(const v8h*)(bRow[j] + k + 16));
}

__device__ __forceinline__ void mm8(v8f acc[2][4], const v16h a[2], const v16h b[4]) {
#pragma unroll
  for (int i = 0; i < 2; ++i)
#pragma unroll
    for (int j = 0; j < 4; ++j)
      acc[i][j] = __builtin_amdgcn_wmma_f32_16x16x32_f16(
          false, a[i], false, b[j], (short)0, acc[i][j], false, false);
}

// ---------------------------------------------------------------------------
// Generic batched WMMA GEMM.
// OUT: 0 = f32 row-major, 1 = f16 row-major, 2 = f16 per-head transposed
//      (m -> (b=m/1024, s=m%1024), n -> (h=n/64, d=n%64),
//       out[((b*16+h)*64+d)*1024 + s])  -- produces V^T tiles directly.
// Block: 256 threads (8 waves), block tile 128x128, wave tile 32x64.
// ---------------------------------------------------------------------------
template<int OUT, bool BIAS, bool RELU, bool CAUSAL, bool MASK>
__global__ __launch_bounds__(256) void wmma_gemm(
    const f16* __restrict__ A, const f16* __restrict__ BT, void* __restrict__ C,
    const float* __restrict__ bias, const float* __restrict__ mask,
    int M, int N, int K, int lda, int ldbt, int ldc,
    long aOut, long aIn, long bOut, long bIn, long cOut, long cIn,
    int zDiv, int maskStride, float alpha)
{
  const int z  = blockIdx.z;
  const int zq = z / zDiv;
  const int zr = z - zq * zDiv;
  A  += (size_t)zq * aOut + (size_t)zr * aIn;
  BT += (size_t)zq * bOut + (size_t)zr * bIn;
  const size_t cOff = (size_t)zq * cOut + (size_t)zr * cIn;
  const float* mrow = MASK ? (mask + (size_t)zq * maskStride) : nullptr;

  const int lane   = threadIdx.x & 31;
  const int wid    = threadIdx.x >> 5;
  const int wm     = wid & 3;     // 4 waves along M
  const int wn     = wid >> 2;    // 2 waves along N
  const int laneLo = lane & 15;
  const int hi     = lane >> 4;   // half-wave selector
  const int cM = blockIdx.y * 128 + wm * 32;
  const int cN = blockIdx.x * 128 + wn * 64;

  const f16* aRow[2];
#pragma unroll
  for (int i = 0; i < 2; ++i) {
    int m = cM + i * 16 + laneLo; if (m >= M) m = 0;   // clamp: result discarded
    aRow[i] = A + (size_t)m * lda + hi * 8;
  }
  const f16* bRow[4];
#pragma unroll
  for (int j = 0; j < 4; ++j) {
    int n = cN + j * 16 + laneLo; if (n >= N) n = 0;
    bRow[j] = BT + (size_t)n * ldbt + hi * 8;
  }

  v8f acc[2][4] = {};

  // Double-buffered main loop (K % 64 == 0).
  v16h a0[2], b0[4], a1[2], b1[4];
  load_frags(aRow, bRow, 0, a0, b0);
  for (int k = 0; k < K; k += 64) {
    __builtin_prefetch(aRow[0] + k + 128, 0, 1);   // global_prefetch, 2 tiles ahead
    __builtin_prefetch(bRow[0] + k + 128, 0, 1);
    load_frags(aRow, bRow, k + 32, a1, b1);        // loads fly under next 8 WMMAs
    mm8(acc, a0, b0);
    const int k2 = (k + 64 < K) ? (k + 64) : 0;    // tail: harmless dummy reload
    load_frags(aRow, bRow, k2, a0, b0);
    mm8(acc, a1, b1);
  }

  float* Cf = (float*)C + cOff;
  f16*   Ch = (f16*)C + cOff;

#pragma unroll
  for (int i = 0; i < 2; ++i)
#pragma unroll
    for (int j = 0; j < 4; ++j)
#pragma unroll
      for (int r = 0; r < 8; ++r) {
        int m = cM + i * 16 + r + hi * 8;
        int n = cN + j * 16 + laneLo;
        if (m < M && n < N) {
          float v = acc[i][j][r] * alpha;
          if (BIAS)   v += bias[n];
          if (CAUSAL) { if (n > m) v -= 1e9f; }
          if (MASK)   v -= 1e9f * mrow[n];
          if (RELU)   v = fmaxf(v, 0.0f);
          if (OUT == 0)      Cf[(size_t)m * ldc + n] = v;
          else if (OUT == 1) Ch[(size_t)m * ldc + n] = (f16)v;
          else {
            int bb = m >> 10, s = m & 1023, h = n >> 6, d = n & 63;
            Ch[(((size_t)bb * 16 + h) * 64 + d) * 1024 + s] = (f16)v;
          }
        }
      }
}

// f32 -> f16 elementwise
__global__ __launch_bounds__(256) void cvt_f32_f16(
    const float* __restrict__ in, f16* __restrict__ out, int n) {
  int i = blockIdx.x * 256 + threadIdx.x;
  if (i < n) out[i] = (f16)in[i];
}

// W[R,C] f32 -> WT[C,R] f16 via LDS tile (bank-conflict padded)
__global__ __launch_bounds__(256) void transpose_cvt(
    const float* __restrict__ in, f16* __restrict__ out, int R, int C) {
  __shared__ float t[32][33];
  const int c0 = blockIdx.x * 32, r0 = blockIdx.y * 32;
  const int tx = threadIdx.x, ty = threadIdx.y;
#pragma unroll
  for (int j = ty; j < 32; j += 8)
    t[j][tx] = in[(size_t)(r0 + j) * C + (c0 + tx)];
  __syncthreads();
#pragma unroll
  for (int j = ty; j < 32; j += 8)
    out[(size_t)(c0 + j) * R + (r0 + tx)] = (f16)t[tx][j];
}

// In-place row softmax over 1024 f16 elements, f32 math. One block per row.
__global__ __launch_bounds__(256) void softmax_rows(f16* __restrict__ s) {
  f16* row = s + (size_t)blockIdx.x * 1024;
  const int t = threadIdx.x, lane = t & 31, wid = t >> 5;
  __shared__ float red[8];
  float v[4], mx = -3.0e38f;
#pragma unroll
  for (int i = 0; i < 4; ++i) { v[i] = (float)row[t + i * 256]; mx = fmaxf(mx, v[i]); }
  mx = warp_max(mx);
  if (lane == 0) red[wid] = mx;
  __syncthreads();
  float m2 = red[0];
#pragma unroll
  for (int i = 1; i < 8; ++i) m2 = fmaxf(m2, red[i]);
  __syncthreads();
  float sum = 0.f;
#pragma unroll
  for (int i = 0; i < 4; ++i) { v[i] = __expf(v[i] - m2); sum += v[i]; }
  sum = warp_sum(sum);
  if (lane == 0) red[wid] = sum;
  __syncthreads();
  float tot = 0.f;
#pragma unroll
  for (int i = 0; i < 8; ++i) tot += red[i];
  const float inv = 1.0f / tot;
#pragma unroll
  for (int i = 0; i < 4; ++i) row[t + i * 256] = (f16)(v[i] * inv);
}

// y = LayerNorm(a + b) over D=1024; writes f32 (and optional f16 copy).
__global__ __launch_bounds__(256) void resid_ln(
    const float* __restrict__ a, const float* __restrict__ b,
    const float* __restrict__ g, const float* __restrict__ be,
    float* __restrict__ o32, f16* __restrict__ o16) {
  const size_t base = (size_t)blockIdx.x * 1024;
  const int t = threadIdx.x, lane = t & 31, wid = t >> 5;
  __shared__ float red[8];
  float v[4], s = 0.f;
#pragma unroll
  for (int i = 0; i < 4; ++i) {
    int idx = t + i * 256;
    v[i] = a[base + idx] + b[base + idx];
    s += v[i];
  }
  s = warp_sum(s);
  if (lane == 0) red[wid] = s;
  __syncthreads();
  float tot = 0.f;
#pragma unroll
  for (int i = 0; i < 8; ++i) tot += red[i];
  __syncthreads();
  const float mean = tot * (1.0f / 1024.0f);
  float vs = 0.f;
#pragma unroll
  for (int i = 0; i < 4; ++i) { float d = v[i] - mean; vs += d * d; }
  vs = warp_sum(vs);
  if (lane == 0) red[wid] = vs;
  __syncthreads();
  float vtot = 0.f;
#pragma unroll
  for (int i = 0; i < 8; ++i) vtot += red[i];
  const float rstd = rsqrtf(vtot * (1.0f / 1024.0f) + 1e-6f);
#pragma unroll
  for (int i = 0; i < 4; ++i) {
    int idx = t + i * 256;
    float y = g[idx] * (v[i] - mean) * rstd + be[idx];
    o32[base + idx] = y;
    if (o16) o16[base + idx] = (f16)y;
  }
}

// ---------------------------------------------------------------------------

extern "C" void kernel_launch(void* const* d_in, const int* in_sizes, int n_in,
                              void* d_out, int out_size, void* d_ws, size_t ws_size,
                              hipStream_t stream) {
  (void)in_sizes; (void)n_in; (void)out_size; (void)ws_size;

  const float* dec   = (const float*)d_in[0];
  const float* enc   = (const float*)d_in[1];
  const float* pmask = (const float*)d_in[2];
  const float* w_sa[4] = {(const float*)d_in[3], (const float*)d_in[4],
                          (const float*)d_in[5], (const float*)d_in[6]};
  const float* w_ca[4] = {(const float*)d_in[7], (const float*)d_in[8],
                          (const float*)d_in[9], (const float*)d_in[10]};
  const float* ff_w1 = (const float*)d_in[11];
  const float* ff_b1 = (const float*)d_in[12];
  const float* ff_w2 = (const float*)d_in[13];
  const float* ff_b2 = (const float*)d_in[14];
  const float* ln1_g = (const float*)d_in[15]; const float* ln1_b = (const float*)d_in[16];
  const float* ln2_g = (const float*)d_in[17]; const float* ln2_b = (const float*)d_in[18];
  const float* ln3_g = (const float*)d_in[19]; const float* ln3_b = (const float*)d_in[20];
  float* outp = (float*)d_out;

  constexpr int  S = 1024, Dm = 1024, Hh = 16, DH = 64, DFF = 4096;
  constexpr int  MALL = 4 * S;                 // B*S = 4096
  constexpr size_t BSD = (size_t)MALL * Dm;    // 4M elements

  char* wp = (char*)d_ws;
  auto alloc = [&](size_t bytes) -> char* {
    char* p = wp; wp += (bytes + 255) & ~(size_t)255; return p;
  };
  f16* x16    = (f16*)alloc(BSD * 2);
  f16* enc16  = (f16*)alloc(BSD * 2);
  f16* wT_sa[4], *wT_ca[4];
  for (int i = 0; i < 4; ++i) wT_sa[i] = (f16*)alloc((size_t)Dm * Dm * 2);
  for (int i = 0; i < 4; ++i) wT_ca[i] = (f16*)alloc((size_t)Dm * Dm * 2);
  f16* w1T    = (f16*)alloc((size_t)Dm * DFF * 2);   // [DFF, Dm]
  f16* w2T    = (f16*)alloc((size_t)Dm * DFF * 2);   // [Dm, DFF]
  f16* q16    = (f16*)alloc(BSD * 2);
  f16* k16    = (f16*)alloc(BSD * 2);
  f16* vT16   = (f16*)alloc(BSD * 2);                // per-head [64, S] tiles
  f16* s16    = (f16*)alloc((size_t)4 * Hh * S * S * 2);   // 128 MB
  f16* o16    = (f16*)alloc(BSD * 2);
  float* attn32 = (float*)alloc(BSD * 4);
  float* y1_32  = (float*)alloc(BSD * 4);
  f16*   y1_16  = (f16*)alloc(BSD * 2);
  float* y2_32  = (float*)alloc(BSD * 4);
  f16*   y2_16  = (f16*)alloc(BSD * 2);
  f16*   h16    = (f16*)alloc((size_t)MALL * DFF * 2);
  float* ff2    = (float*)alloc(BSD * 4);

  const dim3 blk(256);
  const dim3 tblk(32, 8);

  // ---- precision conversion + weight transposes --------------------------
  cvt_f32_f16<<<(int)(BSD / 256), blk, 0, stream>>>(dec, x16, (int)BSD);
  cvt_f32_f16<<<(int)(BSD / 256), blk, 0, stream>>>(enc, enc16, (int)BSD);
  for (int i = 0; i < 4; ++i)
    transpose_cvt<<<dim3(32, 32), tblk, 0, stream>>>(w_sa[i], wT_sa[i], Dm, Dm);
  for (int i = 0; i < 4; ++i)
    transpose_cvt<<<dim3(32, 32), tblk, 0, stream>>>(w_ca[i], wT_ca[i], Dm, Dm);
  transpose_cvt<<<dim3(128, 32), tblk, 0, stream>>>(ff_w1, w1T, Dm, DFF);   // -> [DFF,Dm]
  transpose_cvt<<<dim3(32, 128), tblk, 0, stream>>>(ff_w2, w2T, DFF, Dm);   // -> [Dm,DFF]

  // common grids
  const dim3 gProj(Dm / 128, MALL / 128, 1);       // (8,32)
  const dim3 gScore(S / 128, S / 128, 4 * Hh);     // (8,8,64)
  const dim3 gAV(1, S / 128, 4 * Hh);              // (1,8,64)

  // ========================= self-attention ===============================
  wmma_gemm<1,false,false,false,false><<<gProj, blk, 0, stream>>>(
      x16, wT_sa[0], q16, nullptr, nullptr, MALL, Dm, Dm, Dm, Dm, Dm,
      0,0,0,0,0,0, 1, 0, 1.0f);
  wmma_gemm<1,false,false,false,false><<<gProj, blk, 0, stream>>>(
      x16, wT_sa[1], k16, nullptr, nullptr, MALL, Dm, Dm, Dm, Dm, Dm,
      0,0,0,0,0,0, 1, 0, 1.0f);
  wmma_gemm<2,false,false,false,false><<<gProj, blk, 0, stream>>>(
      x16, wT_sa[2], vT16, nullptr, nullptr, MALL, Dm, Dm, Dm, Dm, Dm,
      0,0,0,0,0,0, 1, 0, 1.0f);

  // scores = (Q K^T)/8 + causal mask, f16, batched over (b,h)
  wmma_gemm<1,false,false,true,false><<<gScore, blk, 0, stream>>>(
      q16, k16, s16, nullptr, nullptr, S, S, DH, Dm, Dm, S,
      (long)S * Dm, DH, (long)S * Dm, DH,
      (long)Hh * S * S, (long)S * S, Hh, 0, 0.125f);
  softmax_rows<<<4 * Hh * S, blk, 0, stream>>>(s16);

  // O_h = P V  -> concat heads (f16)
  wmma_gemm<1,false,false,false,false><<<gAV, blk, 0, stream>>>(
      s16, vT16, o16, nullptr, nullptr, S, DH, S, S, S, Dm,
      (long)Hh * S * S, (long)S * S, (long)Hh * DH * S, (long)DH * S,
      (long)S * Dm, DH, Hh, 0, 1.0f);

  wmma_gemm<0,false,false,false,false><<<gProj, blk, 0, stream>>>(
      o16, wT_sa[3], attn32, nullptr, nullptr, MALL, Dm, Dm, Dm, Dm, Dm,
      0,0,0,0,0,0, 1, 0, 1.0f);
  resid_ln<<<MALL, blk, 0, stream>>>(dec, attn32, ln1_g, ln1_b, y1_32, y1_16);

  // ========================= cross-attention ==============================
  wmma_gemm<1,false,false,false,false><<<gProj, blk, 0, stream>>>(
      y1_16, wT_ca[0], q16, nullptr, nullptr, MALL, Dm, Dm, Dm, Dm, Dm,
      0,0,0,0,0,0, 1, 0, 1.0f);
  wmma_gemm<1,false,false,false,false><<<gProj, blk, 0, stream>>>(
      enc16, wT_ca[1], k16, nullptr, nullptr, MALL, Dm, Dm, Dm, Dm, Dm,
      0,0,0,0,0,0, 1, 0, 1.0f);
  wmma_gemm<2,false,false,false,false><<<gProj, blk, 0, stream>>>(
      enc16, wT_ca[2], vT16, nullptr, nullptr, MALL, Dm, Dm, Dm, Dm, Dm,
      0,0,0,0,0,0, 1, 0, 1.0f);

  wmma_gemm<1,false,false,false,true><<<gScore, blk, 0, stream>>>(
      q16, k16, s16, nullptr, pmask, S, S, DH, Dm, Dm, S,
      (long)S * Dm, DH, (long)S * Dm, DH,
      (long)Hh * S * S, (long)S * S, Hh, S, 0.125f);
  softmax_rows<<<4 * Hh * S, blk, 0, stream>>>(s16);

  wmma_gemm<1,false,false,false,false><<<gAV, blk, 0, stream>>>(
      s16, vT16, o16, nullptr, nullptr, S, DH, S, S, S, Dm,
      (long)Hh * S * S, (long)S * S, (long)Hh * DH * S, (long)DH * S,
      (long)S * Dm, DH, Hh, 0, 1.0f);

  wmma_gemm<0,false,false,false,false><<<gProj, blk, 0, stream>>>(
      o16, wT_ca[3], attn32, nullptr, nullptr, MALL, Dm, Dm, Dm, Dm, Dm,
      0,0,0,0,0,0, 1, 0, 1.0f);
  resid_ln<<<MALL, blk, 0, stream>>>(attn32, y1_32, ln2_g, ln2_b, y2_32, y2_16);

  // ============================== FFN =====================================
  wmma_gemm<1,true,true,false,false><<<dim3(DFF / 128, MALL / 128, 1), blk, 0, stream>>>(
      y2_16, w1T, h16, ff_b1, nullptr, MALL, DFF, Dm, Dm, Dm, DFF,
      0,0,0,0,0,0, 1, 0, 1.0f);
  wmma_gemm<0,true,false,false,false><<<gProj, blk, 0, stream>>>(
      h16, w2T, ff2, ff_b2, nullptr, MALL, Dm, DFF, DFF, DFF, Dm,
      0,0,0,0,0,0, 1, 0, 1.0f);
  resid_ln<<<MALL, blk, 0, stream>>>(y2_32, ff2, ln3_g, ln3_b, outp, nullptr);
}
